// FullAttention1_63462436766089
// MI455X (gfx1250) — compile-verified, matrix operand order checked by reading, not run on hardware
//
#include <hip/hip_runtime.h>
#include <hip/hip_bf16.h>

typedef _Float16 v16h __attribute__((ext_vector_type(16)));
typedef _Float16 h8   __attribute__((ext_vector_type(8)));
typedef float    v8f  __attribute__((ext_vector_type(8)));
typedef unsigned int v4u __attribute__((ext_vector_type(4)));
typedef int      v8i_t __attribute__((ext_vector_type(8)));
typedef int      v4i_t __attribute__((ext_vector_type(4)));

// Problem constants (from reference setup_inputs).
constexpr int Bn  = 2;
constexpr int Qn  = 2048;
constexpr int KN  = 4000;     // total keys
constexpr int Hn  = 8;
constexpr int Dn  = 32;
constexpr int WIN = 100;      // window size
constexpr int NW  = KN / WIN; // 40 windows
constexpr int NWAVE = 4;      // waves (q-tiles) per block
constexpr int KHS = 40;       // Kh row stride (f16): 80B, 16B-aligned, bank-spread
constexpr int VTS = 136;      // Vt/P row stride (f16): 272B, 16B-aligned, bank-spread
constexpr int QTS = 36;       // QTh row stride (f16): 72B, 8B-aligned
constexpr float TEMP = 0.17677669529663687f; // 1/sqrt(32)

constexpr long long PRE_ELEMS = (long long)Bn * Hn * KN * Dn;    // 2,048,000 per tensor
constexpr size_t    WS_NEEDED = (size_t)PRE_ELEMS * 2 * sizeof(_Float16);

// A-layout K index for element e of a v16h operand (16-bit A 16x32 layout):
// lane<16:  K = {0..7, 16..23};  lane>=16: K = {8..15, 24..31}
__device__ __forceinline__ int kOff(int e, int hi) {
    return e + 8 * hi + ((e >= 8) ? 8 : 0);
}

// Concatenate two 8xf16 vectors into a WMMA v16h operand.
__device__ __forceinline__ v16h cat16(h8 a, h8 b) {
    return __builtin_shufflevector(a, b, 0, 1, 2, 3, 4, 5, 6, 7,
                                         8, 9, 10, 11, 12, 13, 14, 15);
}

#if __has_builtin(__builtin_amdgcn_tensor_load_to_lds)
#define HAVE_TDM 1
// Generic 2D TDM load: tile (tile_d0 x tile_d1) from a tensor (tensor_d0 x
// tensor_d1, elem stride0 between dim1 lines), with OOB-zero fill and LDS row
// padding. ds_code: 1 = f16, 2 = f32.
__device__ __forceinline__ void tdm_load_2d(unsigned lds_byte_addr,
                                            unsigned long long gaddr,
                                            unsigned ds_code,
                                            unsigned tensor_d0, unsigned tensor_d1,
                                            unsigned tile_d0, unsigned tile_d1,
                                            unsigned stride0,
                                            unsigned pad_en, unsigned pad_ic,
                                            unsigned pad_ac) {
    v4u g0 = { 1u,
               lds_byte_addr,
               (unsigned)(gaddr & 0xffffffffull),
               (unsigned)((gaddr >> 32) & 0x01ffffffull) | (2u << 30) };
    v8i_t g1 = { (int)((ds_code << 16) | (pad_en << 20) | (pad_ic << 22) | (pad_ac << 25)),
                 (int)(tensor_d0 << 16),
                 (int)((tensor_d0 >> 16) | (tensor_d1 << 16)),
                 (int)((tensor_d1 >> 16) | (tile_d0 << 16)),
                 (int)tile_d1,
                 (int)stride0,
                 0, 0 };
    v4i_t g2 = { 0, 0, 0, 0 };
    v4i_t g3 = { 0, 0, 0, 0 };
    v8i_t g4 = { 0, 0, 0, 0, 0, 0, 0, 0 };  // extra operand on this toolchain
    __builtin_amdgcn_tensor_load_to_lds(g0, g1, g2, g3, g4, 0);
}
#else
#define HAVE_TDM 0
#endif

// ---------------------------------------------------------------------------
// Pre-pass: one-shot f32 -> f16 conversion of K (key-major) and V (transposed,
// dim-major) into workspace, laid out per (b,h) for dense TDM tiles.
//   khg[(bh*KN + k)*Dn + d] = K[b,k,h,d]
//   vtg[(bh*Dn + d)*KN + k] = V[b,k,h,d]
// ---------------------------------------------------------------------------
__global__ void prepass_f16(const float* __restrict__ kg,
                            const float* __restrict__ vg,
                            _Float16* __restrict__ khg,
                            _Float16* __restrict__ vtg)
{
    for (long long i = (long long)blockIdx.x * blockDim.x + threadIdx.x;
         i < PRE_ELEMS; i += (long long)gridDim.x * blockDim.x) {
        // K: i = (bh*KN + k)*Dn + d     (d fastest: coalesced both sides)
        {
            const int d  = (int)(i & 31);
            const long long t1 = i >> 5;
            const int k  = (int)(t1 % KN);
            const int bh = (int)(t1 / KN);
            const int b = bh >> 3, h = bh & 7;
            khg[i] = (_Float16)kg[((size_t)(b * KN + k) * Hn + h) * Dn + d];
        }
        // V: i = (bh*Dn + d)*KN + k     (k fastest: coalesced writes)
        {
            const int k  = (int)(i % KN);
            const long long t2 = i / KN;
            const int d  = (int)(t2 & 31);
            const int bh = (int)(t2 >> 5);
            const int b = bh >> 3, h = bh & 7;
            vtg[i] = (_Float16)vg[((size_t)(b * KN + k) * Hn + h) * Dn + d];
        }
    }
}

// ---------------------------------------------------------------------------
// Main fused kernel, f16 workspace path (TDM window tiles, no conversions).
// ---------------------------------------------------------------------------
__global__ __launch_bounds__(NWAVE * 32) void winattn_f16(
    const float*    __restrict__ qg,    // [B,Q,H,D] f32
    const _Float16* __restrict__ khg,   // [BH,KN,D] f16
    const _Float16* __restrict__ vtg,   // [BH,D,KN] f16
    const float*    __restrict__ wkm,   // [D,D]
    const float*    __restrict__ wvm,   // [D,D]
    float*          __restrict__ outg)  // [B,Q,H,D]
{
    __shared__ _Float16  Kh[112 * KHS];        // K tile f16, rows 100..111 TDM-zeroed
    __shared__ _Float16  Vt[32 * VTS];         // V^T tile f16, key cols 100..127 TDM-zeroed
    __shared__ _Float16  P[NWAVE][16][VTS];    // probs (f16), cols 112..127 = 0
    __shared__ _Float16  QTh[NWAVE][16][QTS];  // queried tile transpose staging

    const int tid  = threadIdx.x;
    const int wv   = tid >> 5;
    const int lane = tid & 31;
    const int lr   = lane & 15;
    const int hi   = lane >> 4;

    const int blocksPerBH = Qn / (16 * NWAVE);   // 32
    const int bh  = blockIdx.x / blocksPerBH;
    const int blk = blockIdx.x % blocksPerBH;
    const int b   = bh / Hn;
    const int h   = bh % Hn;
    const int q0  = (blk * NWAVE + wv) * 16;

    const _Float16* khb = khg + (size_t)bh * KN * Dn;   // [KN][32]
    const _Float16* vtb = vtg + (size_t)bh * Dn * KN;   // [32][KN]

    // ---- Q tile in WMMA A-layout (f16), pre-scaled by softmax temperature ----
    v16h aQ;
    {
        const float* qp = qg + ((size_t)((b * Qn + q0 + lr) * Hn) + h) * Dn;
        #pragma unroll
        for (int e = 0; e < 16; ++e)
            aQ[e] = (_Float16)(qp[kOff(e, hi)] * TEMP);
    }
    // ---- Q tile in C-layout (fp32) for the stage-2 row-dot ----
    float qC0[8], qC1[8];
    #pragma unroll
    for (int r = 0; r < 8; ++r) {
        const float* qp = qg + ((size_t)((b * Qn + q0 + r + 8 * hi) * Hn) + h) * Dn;
        qC0[r] = qp[lr];
        qC1[r] = qp[lr + 16];
    }
    // ---- Projection weights as B operands (B^T per-lane layout): N=e, K=d ----
    v16h bWk0, bWk1, bWv0, bWv1;
    #pragma unroll
    for (int e = 0; e < 16; ++e) {
        const int d = kOff(e, hi);
        bWk0[e] = (_Float16)wkm[lr * Dn + d];
        bWk1[e] = (_Float16)wkm[(lr + 16) * Dn + d];
        bWv0[e] = (_Float16)wvm[lr * Dn + d];
        bWv1[e] = (_Float16)wvm[(lr + 16) * Dn + d];
    }

    // ---- zero static pad regions once ----
    for (int idx = lane; idx < 16 * 16; idx += 32)                 // P cols 112..127
        P[wv][idx >> 4][112 + (idx & 15)] = (_Float16)0.0f;
#if !HAVE_TDM
    for (int idx = tid; idx < 12 * 32; idx += NWAVE * 32)          // Kh rows 100..111
        Kh[(100 + (idx >> 5)) * KHS + (idx & 31)] = (_Float16)0.0f;
    for (int idx = tid; idx < 32 * 28; idx += NWAVE * 32) {        // Vt key cols 100..127
        const int row = idx / 28, col = idx % 28;
        Vt[row * VTS + 100 + col] = (_Float16)0.0f;
    }
#endif

    // ---- stage-2 online-softmax state (per half-wave: m = r + 8*hi) ----
    float Mx[8], Sm[8], oc0[8], oc1[8];
    #pragma unroll
    for (int r = 0; r < 8; ++r) { Mx[r] = -1e30f; Sm[r] = 0.f; oc0[r] = 0.f; oc1[r] = 0.f; }

    for (int w = 0; w < NW; ++w) {
        const int kbase = w * WIN;

        __syncthreads();   // tile buffers free to overwrite

#if HAVE_TDM
        if (wv == 0) {
            // K tile: 112x32 f16, rows >=100 zero-filled; LDS rows padded 32->40 f16
            tdm_load_2d((unsigned)(unsigned long long)(unsigned long)(size_t)&Kh[0],
                        (unsigned long long)(size_t)(khb + (size_t)kbase * Dn),
                        1u, 32u, 100u, 32u, 112u, 32u, 1u, 3u, 3u);
            // V^T tile: 32x128 f16, key cols >=100 zero-filled; rows padded 128->136 f16
            tdm_load_2d((unsigned)(unsigned long long)(unsigned long)(size_t)&Vt[0],
                        (unsigned long long)(size_t)(vtb + kbase),
                        1u, 100u, 32u, 128u, 32u, (unsigned)KN, 1u, 5u, 3u);
            __builtin_amdgcn_s_wait_tensorcnt(0);
        }
#else
        // cooperative f16 copies (pads are statically zeroed above)
        for (int idx = tid; idx < 100 * 4; idx += NWAVE * 32) {
            const int row = idx >> 2, seg = idx & 3;
            *(h8*)&Kh[row * KHS + seg * 8] =
                *(const h8*)(khb + (size_t)(kbase + row) * Dn + seg * 8);
        }
        for (int idx = tid; idx < 32 * 25; idx += NWAVE * 32) {
            const int d = idx / 25, c4 = (idx % 25) * 4;
            #pragma unroll
            for (int j = 0; j < 4; ++j)
                Vt[d * VTS + c4 + j] = vtb[(size_t)d * KN + kbase + c4 + j];
        }
#endif
        __syncthreads();   // tiles ready for all waves

        // ---- stage 1: QK^T over 7 key tiles of 16 (scores stay in regs) ----
        v8f ct[7];
        #pragma unroll
        for (int t = 0; t < 7; ++t) {
            const int krow = t * 16 + lr;          // 0..111 (rows 100..111 zero)
            const _Float16* kr = &Kh[krow * KHS + 8 * hi];
            const v16h bK = cat16(*(const h8*)kr, *(const h8*)(kr + 16));
            v8f cz = {};
            ct[t] = __builtin_amdgcn_wmma_f32_16x16x32_f16(false, aQ, false, bK,
                                                           (short)0, cz, false, false);
        }

        // ---- register softmax over the 100 window keys ----
        float mx[8];
        #pragma unroll
        for (int r = 0; r < 8; ++r) mx[r] = -1e30f;
        #pragma unroll
        for (int t = 0; t < 7; ++t) {
            const bool valid = (t < 6) | (lr < 4);
            #pragma unroll
            for (int r = 0; r < 8; ++r) {
                ct[t][r] = valid ? ct[t][r] : -1e30f;
                mx[r] = fmaxf(mx[r], ct[t][r]);
            }
        }
        #pragma unroll
        for (int mk = 1; mk <= 8; mk <<= 1) {
            #pragma unroll
            for (int r = 0; r < 8; ++r)
                mx[r] = fmaxf(mx[r], __shfl_xor(mx[r], mk));
        }
        float sum[8];
        #pragma unroll
        for (int r = 0; r < 8; ++r) sum[r] = 0.f;
        #pragma unroll
        for (int t = 0; t < 7; ++t) {
            #pragma unroll
            for (int r = 0; r < 8; ++r) {
                const float p = __expf(ct[t][r] - mx[r]);  // masked -> exp(-inf)=0
                ct[t][r] = p;
                sum[r] += p;
            }
        }
        #pragma unroll
        for (int mk = 1; mk <= 8; mk <<= 1) {
            #pragma unroll
            for (int r = 0; r < 8; ++r)
                sum[r] += __shfl_xor(sum[r], mk);
        }
        #pragma unroll
        for (int r = 0; r < 8; ++r) sum[r] = 1.0f / sum[r];
        // store UNNORMALIZED probs; 1/sum applied to queried after the matmul
        #pragma unroll
        for (int t = 0; t < 7; ++t) {
            #pragma unroll
            for (int r = 0; r < 8; ++r)
                P[wv][r + 8 * hi][t * 16 + lr] = (_Float16)ct[t][r];
        }
        __syncthreads();   // publish P across lanes

        // ---- queried = P(16x128) x V(128x32): 4 K-chunks x 2 dim tiles ----
        v8f qd0 = {}, qd1 = {};
        const _Float16* Prow = &P[wv][lr][0];
        const _Float16* Vra  = &Vt[lr * VTS];
        const _Float16* Vrb  = &Vt[(lr + 16) * VTS];
        #pragma unroll
        for (int ch = 0; ch < 4; ++ch) {
            const int base = ch * 32 + 8 * hi;
            const v16h aP  = cat16(*(const h8*)(Prow + base), *(const h8*)(Prow + base + 16));
            const v16h bVa = cat16(*(const h8*)(Vra + base),  *(const h8*)(Vra + base + 16));
            const v16h bVb = cat16(*(const h8*)(Vrb + base),  *(const h8*)(Vrb + base + 16));
            qd0 = __builtin_amdgcn_wmma_f32_16x16x32_f16(false, aP, false, bVa,
                                                         (short)0, qd0, false, false);
            qd1 = __builtin_amdgcn_wmma_f32_16x16x32_f16(false, aP, false, bVb,
                                                         (short)0, qd1, false, false);
        }
        #pragma unroll
        for (int r = 0; r < 8; ++r) { qd0[r] *= sum[r]; qd1[r] *= sum[r]; }

        // ---- transpose queried through LDS into A-layout (f16) ----
        #pragma unroll
        for (int r = 0; r < 8; ++r) {
            QTh[wv][r + 8 * hi][lr]      = (_Float16)qd0[r];
            QTh[wv][r + 8 * hi][lr + 16] = (_Float16)qd1[r];
        }
        __syncthreads();
        v16h aQd;
        #pragma unroll
        for (int e = 0; e < 16; ++e)
            aQd[e] = QTh[wv][lr][kOff(e, hi)];

        // ---- projections: key_in / value_in (queried x W^T) ----
        v8f cz = {};
        v8f ki0 = __builtin_amdgcn_wmma_f32_16x16x32_f16(false, aQd, false, bWk0, (short)0, cz, false, false);
        v8f ki1 = __builtin_amdgcn_wmma_f32_16x16x32_f16(false, aQd, false, bWk1, (short)0, cz, false, false);
        v8f vi0 = __builtin_amdgcn_wmma_f32_16x16x32_f16(false, aQd, false, bWv0, (short)0, cz, false, false);
        v8f vi1 = __builtin_amdgcn_wmma_f32_16x16x32_f16(false, aQd, false, bWv1, (short)0, cz, false, false);

        // ---- stage-2 score: per-row dot q . key_in (reduce across 16 lanes) ----
        float part[8];
        #pragma unroll
        for (int r = 0; r < 8; ++r)
            part[r] = qC0[r] * ki0[r] + qC1[r] * ki1[r];
        #pragma unroll
        for (int mk = 1; mk <= 8; mk <<= 1) {
            #pragma unroll
            for (int r = 0; r < 8; ++r)
                part[r] += __shfl_xor(part[r], mk);
        }

        // ---- online softmax over windows + accumulate value_in ----
        #pragma unroll
        for (int r = 0; r < 8; ++r) {
            const float s  = part[r] * TEMP;
            const float nm = fmaxf(Mx[r], s);
            const float al = __expf(Mx[r] - nm);
            const float p  = __expf(s - nm);
            Sm[r]  = Sm[r] * al + p;
            oc0[r] = oc0[r] * al + p * vi0[r];
            oc1[r] = oc1[r] * al + p * vi1[r];
            Mx[r]  = nm;
        }
    }

    // ---- writeback: out[b, q0+m, h, :] ----
    #pragma unroll
    for (int r = 0; r < 8; ++r) {
        const int m = r + 8 * hi;
        const float inv = 1.0f / Sm[r];
        float* op = outg + ((size_t)((b * Qn + q0 + m) * Hn) + h) * Dn;
        op[lr]      = oc0[r] * inv;
        op[lr + 16] = oc1[r] * inv;
    }
}

// ---------------------------------------------------------------------------
// Legacy fallback (no workspace): Round-4 kernel, f32 staging + in-loop convert.
// ---------------------------------------------------------------------------
__global__ __launch_bounds__(NWAVE * 32) void winattn_legacy(
    const float* __restrict__ qg, const float* __restrict__ kg,
    const float* __restrict__ vg, const float* __restrict__ wkm,
    const float* __restrict__ wvm, float* __restrict__ outg)
{
    __shared__ float     Kst[WIN * 32];
    __shared__ float     Vst[WIN * 32];
    __shared__ _Float16  Kh[112 * KHS];
    __shared__ _Float16  Vt[32 * VTS];
    __shared__ _Float16  P[NWAVE][16][VTS];
    __shared__ _Float16  QTh[NWAVE][16][QTS];

    const int tid  = threadIdx.x;
    const int wv   = tid >> 5;
    const int lane = tid & 31;
    const int lr   = lane & 15;
    const int hi   = lane >> 4;

    const int blocksPerBH = Qn / (16 * NWAVE);
    const int bh  = blockIdx.x / blocksPerBH;
    const int blk = blockIdx.x % blocksPerBH;
    const int b   = bh / Hn;
    const int h   = bh % Hn;
    const int q0  = (blk * NWAVE + wv) * 16;

    const size_t kvstride = (size_t)Hn * Dn;
    const float* kb = kg + ((size_t)b * KN * Hn + h) * Dn;
    const float* vb = vg + ((size_t)b * KN * Hn + h) * Dn;

    v16h aQ;
    {
        const float* qp = qg + ((size_t)((b * Qn + q0 + lr) * Hn) + h) * Dn;
        #pragma unroll
        for (int e = 0; e < 16; ++e)
            aQ[e] = (_Float16)(qp[kOff(e, hi)] * TEMP);
    }
    float qC0[8], qC1[8];
    #pragma unroll
    for (int r = 0; r < 8; ++r) {
        const float* qp = qg + ((size_t)((b * Qn + q0 + r + 8 * hi) * Hn) + h) * Dn;
        qC0[r] = qp[lr];
        qC1[r] = qp[lr + 16];
    }
    v16h bWk0, bWk1, bWv0, bWv1;
    #pragma unroll
    for (int e = 0; e < 16; ++e) {
        const int d = kOff(e, hi);
        bWk0[e] = (_Float16)wkm[lr * Dn + d];
        bWk1[e] = (_Float16)wkm[(lr + 16) * Dn + d];
        bWv0[e] = (_Float16)wvm[lr * Dn + d];
        bWv1[e] = (_Float16)wvm[(lr + 16) * Dn + d];
    }

    for (int idx = tid; idx < 12 * 32; idx += NWAVE * 32)
        Kh[(100 + (idx >> 5)) * KHS + (idx & 31)] = (_Float16)0.0f;
    for (int idx = tid; idx < 32 * 28; idx += NWAVE * 32) {
        const int row = idx / 28, col = idx % 28;
        Vt[row * VTS + 100 + col] = (_Float16)0.0f;
    }
    for (int idx = lane; idx < 16 * 16; idx += 32)
        P[wv][idx >> 4][112 + (idx & 15)] = (_Float16)0.0f;

    float Mx[8], Sm[8], oc0[8], oc1[8];
    #pragma unroll
    for (int r = 0; r < 8; ++r) { Mx[r] = -1e30f; Sm[r] = 0.f; oc0[r] = 0.f; oc1[r] = 0.f; }

    for (int w = 0; w < NW; ++w) {
        const int kbase = w * WIN;
        __syncthreads();
#if HAVE_TDM
        if (wv == 0) {
            tdm_load_2d((unsigned)(unsigned long long)(unsigned long)(size_t)&Kst[0],
                        (unsigned long long)(size_t)(kb + (size_t)kbase * kvstride),
                        2u, 32u, 100u, 32u, 100u, 256u, 0u, 0u, 0u);
            tdm_load_2d((unsigned)(unsigned long long)(unsigned long)(size_t)&Vst[0],
                        (unsigned long long)(size_t)(vb + (size_t)kbase * kvstride),
                        2u, 32u, 100u, 32u, 100u, 256u, 0u, 0u, 0u);
            __builtin_amdgcn_s_wait_tensorcnt(0);
        }
#else
        for (int idx = tid; idx < WIN * 8; idx += NWAVE * 32) {
            const int row = idx >> 3, seg = idx & 7;
            *(float4*)&Kst[row * 32 + seg * 4] =
                *(const float4*)(kb + (size_t)(kbase + row) * kvstride + seg * 4);
            *(float4*)&Vst[row * 32 + seg * 4] =
                *(const float4*)(vb + (size_t)(kbase + row) * kvstride + seg * 4);
        }
#endif
        __syncthreads();
        for (int idx = tid; idx < WIN * 8; idx += NWAVE * 32) {
            const int row = idx >> 3, c4 = (idx & 7) * 4;
            const float4 kk4 = *(const float4*)&Kst[row * 32 + c4];
            Kh[row * KHS + c4 + 0] = (_Float16)kk4.x;
            Kh[row * KHS + c4 + 1] = (_Float16)kk4.y;
            Kh[row * KHS + c4 + 2] = (_Float16)kk4.z;
            Kh[row * KHS + c4 + 3] = (_Float16)kk4.w;
            const float4 vv4 = *(const float4*)&Vst[row * 32 + c4];
            Vt[(c4 + 0) * VTS + row] = (_Float16)vv4.x;
            Vt[(c4 + 1) * VTS + row] = (_Float16)vv4.y;
            Vt[(c4 + 2) * VTS + row] = (_Float16)vv4.z;
            Vt[(c4 + 3) * VTS + row] = (_Float16)vv4.w;
        }
        __syncthreads();

        v8f ct[7];
        #pragma unroll
        for (int t = 0; t < 7; ++t) {
            const int krow = t * 16 + lr;
            const _Float16* kr = &Kh[krow * KHS + 8 * hi];
            const v16h bK = cat16(*(const h8*)kr, *(const h8*)(kr + 16));
            v8f cz = {};
            ct[t] = __builtin_amdgcn_wmma_f32_16x16x32_f16(false, aQ, false, bK,
                                                           (short)0, cz, false, false);
        }
        float mx[8];
        #pragma unroll
        for (int r = 0; r < 8; ++r) mx[r] = -1e30f;
        #pragma unroll
        for (int t = 0; t < 7; ++t) {
            const bool valid = (t < 6) | (lr < 4);
            #pragma unroll
            for (int r = 0; r < 8; ++r) {
                ct[t][r] = valid ? ct[t][r] : -1e30f;
                mx[r] = fmaxf(mx[r], ct[t][r]);
            }
        }
        #pragma unroll
        for (int mk = 1; mk <= 8; mk <<= 1)
            #pragma unroll
            for (int r = 0; r < 8; ++r)
                mx[r] = fmaxf(mx[r], __shfl_xor(mx[r], mk));
        float sum[8];
        #pragma unroll
        for (int r = 0; r < 8; ++r) sum[r] = 0.f;
        #pragma unroll
        for (int t = 0; t < 7; ++t)
            #pragma unroll
            for (int r = 0; r < 8; ++r) {
                const float p = __expf(ct[t][r] - mx[r]);
                ct[t][r] = p;
                sum[r] += p;
            }
        #pragma unroll
        for (int mk = 1; mk <= 8; mk <<= 1)
            #pragma unroll
            for (int r = 0; r < 8; ++r)
                sum[r] += __shfl_xor(sum[r], mk);
        #pragma unroll
        for (int r = 0; r < 8; ++r) sum[r] = 1.0f / sum[r];
        #pragma unroll
        for (int t = 0; t < 7; ++t)
            #pragma unroll
            for (int r = 0; r < 8; ++r)
                P[wv][r + 8 * hi][t * 16 + lr] = (_Float16)ct[t][r];
        __syncthreads();

        v8f qd0 = {}, qd1 = {};
        const _Float16* Prow = &P[wv][lr][0];
        const _Float16* Vra  = &Vt[lr * VTS];
        const _Float16* Vrb  = &Vt[(lr + 16) * VTS];
        #pragma unroll
        for (int ch = 0; ch < 4; ++ch) {
            const int base = ch * 32 + 8 * hi;
            const v16h aP  = cat16(*(const h8*)(Prow + base), *(const h8*)(Prow + base + 16));
            const v16h bVa = cat16(*(const h8*)(Vra + base),  *(const h8*)(Vra + base + 16));
            const v16h bVb = cat16(*(const h8*)(Vrb + base),  *(const h8*)(Vrb + base + 16));
            qd0 = __builtin_amdgcn_wmma_f32_16x16x32_f16(false, aP, false, bVa,
                                                         (short)0, qd0, false, false);
            qd1 = __builtin_amdgcn_wmma_f32_16x16x32_f16(false, aP, false, bVb,
                                                         (short)0, qd1, false, false);
        }
        #pragma unroll
        for (int r = 0; r < 8; ++r) { qd0[r] *= sum[r]; qd1[r] *= sum[r]; }

        #pragma unroll
        for (int r = 0; r < 8; ++r) {
            QTh[wv][r + 8 * hi][lr]      = (_Float16)qd0[r];
            QTh[wv][r + 8 * hi][lr + 16] = (_Float16)qd1[r];
        }
        __syncthreads();
        v16h aQd;
        #pragma unroll
        for (int e = 0; e < 16; ++e)
            aQd[e] = QTh[wv][lr][kOff(e, hi)];

        v8f cz = {};
        v8f ki0 = __builtin_amdgcn_wmma_f32_16x16x32_f16(false, aQd, false, bWk0, (short)0, cz, false, false);
        v8f ki1 = __builtin_amdgcn_wmma_f32_16x16x32_f16(false, aQd, false, bWk1, (short)0, cz, false, false);
        v8f vi0 = __builtin_amdgcn_wmma_f32_16x16x32_f16(false, aQd, false, bWv0, (short)0, cz, false, false);
        v8f vi1 = __builtin_amdgcn_wmma_f32_16x16x32_f16(false, aQd, false, bWv1, (short)0, cz, false, false);

        float part[8];
        #pragma unroll
        for (int r = 0; r < 8; ++r)
            part[r] = qC0[r] * ki0[r] + qC1[r] * ki1[r];
        #pragma unroll
        for (int mk = 1; mk <= 8; mk <<= 1)
            #pragma unroll
            for (int r = 0; r < 8; ++r)
                part[r] += __shfl_xor(part[r], mk);

        #pragma unroll
        for (int r = 0; r < 8; ++r) {
            const float s  = part[r] * TEMP;
            const float nm = fmaxf(Mx[r], s);
            const float al = __expf(Mx[r] - nm);
            const float p  = __expf(s - nm);
            Sm[r]  = Sm[r] * al + p;
            oc0[r] = oc0[r] * al + p * vi0[r];
            oc1[r] = oc1[r] * al + p * vi1[r];
            Mx[r]  = nm;
        }
    }

    #pragma unroll
    for (int r = 0; r < 8; ++r) {
        const int m = r + 8 * hi;
        const float inv = 1.0f / Sm[r];
        float* op = outg + ((size_t)((b * Qn + q0 + m) * Hn) + h) * Dn;
        op[lr]      = oc0[r] * inv;
        op[lr + 16] = oc1[r] * inv;
    }
}

extern "C" void kernel_launch(void* const* d_in, const int* in_sizes, int n_in,
                              void* d_out, int out_size, void* d_ws, size_t ws_size,
                              hipStream_t stream) {
    (void)in_sizes; (void)n_in; (void)out_size;
    const float* qg  = (const float*)d_in[0];
    const float* kg  = (const float*)d_in[1];
    const float* vg  = (const float*)d_in[2];
    const float* wkm = (const float*)d_in[3];
    const float* wvm = (const float*)d_in[4];
    float* outg = (float*)d_out;

    const int blocks = Bn * Hn * (Qn / (16 * NWAVE)); // 512 blocks x 128 threads

    if (d_ws != nullptr && ws_size >= WS_NEEDED) {
        _Float16* khg = (_Float16*)d_ws;
        _Float16* vtg = khg + PRE_ELEMS;
        prepass_f16<<<2048, 256, 0, stream>>>(kg, vg, khg, vtg);
        winattn_f16<<<blocks, NWAVE * 32, 0, stream>>>(qg, khg, vtg, wkm, wvm, outg);
    } else {
        winattn_legacy<<<blocks, NWAVE * 32, 0, stream>>>(qg, kg, vg, wkm, wvm, outg);
    }
}
